// FedIGL_GIN_63634235457722
// MI455X (gfx1250) — matrix operand order repaired and checked by reading, not compiled, not run on hardware
//
#include <hip/hip_runtime.h>
#include <hip/hip_bf16.h>

typedef __bf16 bf16_t;
typedef __attribute__((ext_vector_type(16))) __bf16 v16bf;
typedef __attribute__((ext_vector_type(8)))  __bf16 v8bf;
typedef __attribute__((ext_vector_type(8)))  float  v8f;

#define NN    50000
#define EE    800000
#define NF    64
#define NH    128
#define NG    128
#define NCLS  10
#define KTOP  400000
#define LIDX  2          // only last layer of each stack matters (reference overwrites)
#define TIE_CHUNKS 2048

static __device__ __forceinline__ int imin(int a, int b) { return a < b ? a : b; }

// ---------------------------------------------------------------------------
// WMMA strip GEMM: out = act(A[MxK] @ Wt^T + bias), Wt stored [NOUT x K].
// 256-thread block = 8 waves; Wt staged once into LDS; each wave owns a
// 16 x (NT*16) output strip: A fragment loaded once per k-step, reused across
// all NT column tiles (8 WMMAs per A fetch in steady state, B from ds_load).
// ---------------------------------------------------------------------------
template <int NT, int K>
__global__ __launch_bounds__(256) void gemm_wmma_strip(
    const bf16_t* __restrict__ A, const bf16_t* __restrict__ Wt,
    const float* __restrict__ bias, float* __restrict__ outF,
    bf16_t* __restrict__ outB, int Mtiles, int relu)
{
  constexpr int NOUT = NT * 16;
  __shared__ bf16_t s_wt[NOUT * K];
  constexpr int VECS = (NOUT * K) / 8;               // 8 bf16 per uint4
  for (int i = threadIdx.x; i < VECS; i += 256)
    ((uint4*)s_wt)[i] = ((const uint4*)Wt)[i];
  __syncthreads();

  const int tM = blockIdx.x * 8 + (threadIdx.x >> 5);
  if (tM >= Mtiles) return;                          // wave-uniform: EXEC stays all-1
  const int lane = threadIdx.x & 31;
  const int half = lane >> 4, l16 = lane & 15;

  const bf16_t* Arow = A + (size_t)(tM * 16 + l16) * K;
  v8f acc[NT];
  const v8f zero = {0.f, 0.f, 0.f, 0.f, 0.f, 0.f, 0.f, 0.f};
#pragma unroll
  for (int t = 0; t < NT; ++t) acc[t] = zero;

#pragma unroll
  for (int kb = 0; kb < K; kb += 32) {
    union { v16bf v; v8bf h[2]; } a;
    a.h[0] = *(const v8bf*)(Arow + kb + 16 * half);      // K = kb+16*half .. +7
    a.h[1] = *(const v8bf*)(Arow + kb + 16 * half + 8);  // K = kb+16*half+8 .. +15
#pragma unroll
    for (int t = 0; t < NT; ++t) {
      v16bf b = *(const v16bf*)(s_wt + (size_t)(t * 16 + l16) * K + kb + 16 * half);
      acc[t] = __builtin_amdgcn_wmma_f32_16x16x32_bf16(false, a.v, false, b,
                                                       (short)0, acc[t], false, false);
    }
  }

#pragma unroll
  for (int t = 0; t < NT; ++t) {
    const int n = t * 16 + l16;
    const float bv = bias ? bias[n] : 0.0f;
#pragma unroll
    for (int v = 0; v < 8; ++v) {
      const int m = tM * 16 + v + 8 * half;   // C layout: VGPR v holds M=v / M=v+8
      float val = acc[t][v] + bv;
      if (relu) val = fmaxf(val, 0.0f);
      if (outF) outF[(size_t)m * NOUT + n] = val;
      if (outB) outB[(size_t)m * NOUT + n] = (bf16_t)val;
    }
  }
}

// ---------------------------------------------------------------------------
// Elementwise / conversion helpers
// ---------------------------------------------------------------------------
__global__ void k_zero_f32(float* p, long n) {
  long i = (long)blockIdx.x * blockDim.x + threadIdx.x;
  if (i < n) p[i] = 0.0f;
}
__global__ void k_f32_to_bf16(const float* __restrict__ in, bf16_t* __restrict__ out, long n) {
  long i = (long)blockIdx.x * blockDim.x + threadIdx.x;
  if (i < n) out[i] = (bf16_t)in[i];
}
// Wt[n*K + k] = (n < Nin ? W[k*Nin + n] : 0)   (W row-major [K x Nin])
__global__ void k_transpose_bf16(const float* __restrict__ W, bf16_t* __restrict__ Wt,
                                 int K, int Nin, int Npad) {
  int idx = blockIdx.x * blockDim.x + threadIdx.x;
  if (idx >= Npad * K) return;
  int n = idx / K, k = idx - n * K;
  Wt[idx] = (bf16_t)((n < Nin) ? W[(size_t)k * Nin + n] : 0.0f);
}
__global__ void k_pad_bias(const float* __restrict__ b, float* __restrict__ bp, int n, int npad) {
  int i = blockIdx.x * blockDim.x + threadIdx.x;
  if (i < npad) bp[i] = (i < n) ? b[i] : 0.0f;
}
// out_bf16 = bf16(agg + 2*h)   (GIN: self-loop + (1+eps)*x, eps=0)
__global__ void k_add2h_bf16(const float* __restrict__ agg, const float* __restrict__ h,
                             bf16_t* __restrict__ out, long n) {
  long i = (long)blockIdx.x * blockDim.x + threadIdx.x;
  if (i < n) out[i] = (bf16_t)(agg[i] + 2.0f * h[i]);
}

// ---------------------------------------------------------------------------
// Edge scatter: one wave per edge, 4 feats per lane (128 total)
// ---------------------------------------------------------------------------
__global__ __launch_bounds__(256) void k_scatter_ones(
    const float* __restrict__ h, const int* __restrict__ row, const int* __restrict__ col,
    float* __restrict__ agg) {
  int w = (blockIdx.x * 256 + threadIdx.x) >> 5;
  int lane = threadIdx.x & 31;
  if (w >= EE) return;
  int r = row[w], c = col[w];
  float4 v = *(const float4*)(h + (size_t)r * NH + lane * 4);
  float* dst = agg + (size_t)c * NH + lane * 4;
  atomicAdd(dst + 0, v.x); atomicAdd(dst + 1, v.y);
  atomicAdd(dst + 2, v.z); atomicAdd(dst + 3, v.w);
}
__global__ __launch_bounds__(256) void k_scatter_split(
    const float* __restrict__ h, const int* __restrict__ row, const int* __restrict__ col,
    const float* __restrict__ ew, const unsigned* __restrict__ flag,
    float* __restrict__ agg1, float* __restrict__ agg2) {
  int w = (blockIdx.x * 256 + threadIdx.x) >> 5;
  int lane = threadIdx.x & 31;
  if (w >= EE) return;
  int r = row[w], c = col[w];
  float wt = ew[w];
  float* dst = (flag[w] ? agg1 : agg2) + (size_t)c * NH + lane * 4;
  float4 v = *(const float4*)(h + (size_t)r * NH + lane * 4);
  atomicAdd(dst + 0, wt * v.x); atomicAdd(dst + 1, wt * v.y);
  atomicAdd(dst + 2, wt * v.z); atomicAdd(dst + 3, wt * v.w);
}

// ---------------------------------------------------------------------------
// Per-edge dot of inv_x rows -> ew + monotone sortable key
// ---------------------------------------------------------------------------
__global__ __launch_bounds__(256) void k_edge_dot(
    const float* __restrict__ inv_x, const int* __restrict__ row, const int* __restrict__ col,
    float* __restrict__ ew, unsigned* __restrict__ key) {
  int w = (blockIdx.x * 256 + threadIdx.x) >> 5;
  int lane = threadIdx.x & 31;
  if (w >= EE) return;
  int r = row[w], c = col[w];
  float4 a = *(const float4*)(inv_x + (size_t)r * NH + lane * 4);
  float4 b = *(const float4*)(inv_x + (size_t)c * NH + lane * 4);
  float s = a.x * b.x + a.y * b.y + a.z * b.z + a.w * b.w;
#pragma unroll
  for (int off = 16; off > 0; off >>= 1) s += __shfl_xor(s, off, 32);
  if (lane == 0) {
    ew[w] = s;
    unsigned u = __float_as_uint(s);
    key[w] = (u & 0x80000000u) ? ~u : (u | 0x80000000u);
  }
}

// ---------------------------------------------------------------------------
// Radix select of the KTOP-th largest key (4 passes, MSB first).
// state[0]=prefix, state[1]=rank still needed inside current bin.
// ---------------------------------------------------------------------------
__global__ void k_sel_init(unsigned* state, unsigned* hist) {
  if (threadIdx.x < 256) hist[threadIdx.x] = 0u;
  if (threadIdx.x == 0) { state[0] = 0u; state[1] = (unsigned)KTOP; }
}
__global__ __launch_bounds__(256) void k_radix_hist(
    const unsigned* __restrict__ key, unsigned* __restrict__ hist,
    const unsigned* __restrict__ state, int shift) {
  __shared__ unsigned sh[256];
  sh[threadIdx.x] = 0u;
  __syncthreads();
  unsigned pref = state[0];
  for (int i = blockIdx.x * 256 + threadIdx.x; i < EE; i += gridDim.x * 256) {
    unsigned kk = key[i];
    bool match = (shift == 24) || ((kk >> (shift + 8)) == pref);
    if (match) atomicAdd(&sh[(kk >> shift) & 255u], 1u);
  }
  __syncthreads();
  if (sh[threadIdx.x]) atomicAdd(&hist[threadIdx.x], sh[threadIdx.x]);
}
__global__ void k_radix_select(unsigned* hist, unsigned* state) {
  if (threadIdx.x != 0 || blockIdx.x != 0) return;
  unsigned needed = state[1];
  int b = 255;
  for (;; --b) {
    unsigned cnt = hist[b];
    if (cnt >= needed) break;
    needed -= cnt;
    if (b == 0) break;
  }
  state[0] = (state[0] << 8) | (unsigned)b;
  state[1] = needed;
  for (int i = 0; i < 256; ++i) hist[i] = 0u;
}

// ---------------------------------------------------------------------------
// Flagging: key > T -> top; ties (key == T) taken lowest-index-first until
// `needed` filled (matches jax.lax.top_k tie behavior).
// ---------------------------------------------------------------------------
__global__ void k_flag_strict(const unsigned* __restrict__ key,
                              const unsigned* __restrict__ state,
                              unsigned* __restrict__ flag) {
  int i = blockIdx.x * blockDim.x + threadIdx.x;
  if (i >= EE) return;
  flag[i] = (key[i] > state[0]) ? 1u : 0u;
}
__global__ void k_tie_count(const unsigned* __restrict__ key, const unsigned* __restrict__ state,
                            unsigned* __restrict__ tcnt, int chunk) {
  int t = blockIdx.x * blockDim.x + threadIdx.x;
  if (t >= TIE_CHUNKS) return;
  unsigned T = state[0];
  int s = t * chunk, e = imin(EE, s + chunk);
  unsigned c = 0;
  for (int i = s; i < e; ++i) c += (key[i] == T) ? 1u : 0u;
  tcnt[t] = c;
}
__global__ void k_tie_scan(unsigned* tcnt) {
  if (threadIdx.x != 0 || blockIdx.x != 0) return;
  unsigned acc = 0;
  for (int t = 0; t < TIE_CHUNKS; ++t) { unsigned c = tcnt[t]; tcnt[t] = acc; acc += c; }
}
__global__ void k_tie_flag(const unsigned* __restrict__ key, const unsigned* __restrict__ state,
                           const unsigned* __restrict__ tcnt, unsigned* __restrict__ flag,
                           int chunk) {
  int t = blockIdx.x * blockDim.x + threadIdx.x;
  if (t >= TIE_CHUNKS) return;
  unsigned T = state[0], needed = state[1];
  unsigned r = tcnt[t];
  int s = t * chunk, e = imin(EE, s + chunk);
  for (int i = s; i < e; ++i)
    if (key[i] == T) { flag[i] = (r < needed) ? 1u : 0u; ++r; }
}

// ---------------------------------------------------------------------------
// Pool + log-softmax
// ---------------------------------------------------------------------------
__global__ void k_pool(const float* __restrict__ x1, const float* __restrict__ x2,
                       const int* __restrict__ batch, float* __restrict__ pooled) {
  long i = (long)blockIdx.x * blockDim.x + threadIdx.x;
  if (i >= (long)NN * NH) return;
  int node = (int)(i >> 7), f = (int)(i & 127);
  atomicAdd(&pooled[(size_t)batch[node] * NH + f], x1[i] + x2[i]);
}
__global__ void k_log_softmax(const float* __restrict__ logits, float* __restrict__ out) {
  int r = blockIdx.x * blockDim.x + threadIdx.x;
  if (r >= NG) return;
  float m = -1e30f;
  for (int c = 0; c < NCLS; ++c) m = fmaxf(m, logits[r * 16 + c]);
  float s = 0.0f;
  for (int c = 0; c < NCLS; ++c) s += expf(logits[r * 16 + c] - m);
  float ls = logf(s);
  for (int c = 0; c < NCLS; ++c) out[r * NCLS + c] = logits[r * 16 + c] - m - ls;
}

// ---------------------------------------------------------------------------
// Orchestration
// ---------------------------------------------------------------------------
static char* bump(char*& p, size_t bytes) {
  char* r = p;
  p += (bytes + 255) & ~(size_t)255;
  return r;
}

extern "C" void kernel_launch(void* const* d_in, const int* in_sizes, int n_in,
                              void* d_out, int out_size, void* d_ws, size_t ws_size,
                              hipStream_t stream) {
  (void)in_sizes; (void)n_in; (void)out_size; (void)ws_size;
  const float* x      = (const float*)d_in[0];
  const int*   eidx   = (const int*)d_in[1];
  const int*   row    = eidx;
  const int*   col    = eidx + EE;
  const int*   batch  = (const int*)d_in[2];
  const float* pre_W  = (const float*)d_in[3];
  const float* pre_b  = (const float*)d_in[4];
  const float* c1_W1  = (const float*)d_in[5]  + (size_t)LIDX * NH * NH;
  const float* c1_b1  = (const float*)d_in[6]  + (size_t)LIDX * NH;
  const float* c1_W2  = (const float*)d_in[7]  + (size_t)LIDX * NH * NH;
  const float* c1_b2  = (const float*)d_in[8]  + (size_t)LIDX * NH;
  const float* c3_W1  = (const float*)d_in[9]  + (size_t)LIDX * NH * NH;
  const float* c3_b1  = (const float*)d_in[10] + (size_t)LIDX * NH;
  const float* c3_W2  = (const float*)d_in[11] + (size_t)LIDX * NH * NH;
  const float* c3_b2  = (const float*)d_in[12] + (size_t)LIDX * NH;
  const float* c4_W1  = (const float*)d_in[13] + (size_t)LIDX * NH * NH;
  const float* c4_b1  = (const float*)d_in[14] + (size_t)LIDX * NH;
  const float* c4_W2  = (const float*)d_in[15] + (size_t)LIDX * NH * NH;
  const float* c4_b2  = (const float*)d_in[16] + (size_t)LIDX * NH;
  const float* post_W = (const float*)d_in[17];
  const float* post_b = (const float*)d_in[18];
  const float* ro_W   = (const float*)d_in[19];
  const float* ro_b   = (const float*)d_in[20];

  char* ws = (char*)d_ws;
  float*  h       = (float*)bump(ws, (size_t)NN * NH * 4);
  bf16_t* h_bf    = (bf16_t*)bump(ws, (size_t)NN * NH * 2);
  bf16_t* x_bf    = (bf16_t*)bump(ws, (size_t)NN * NF * 2);
  float*  inv_x   = (float*)bump(ws, (size_t)NN * NH * 4);
  float*  agg1    = (float*)bump(ws, (size_t)NN * NH * 4);
  float*  agg2    = (float*)bump(ws, (size_t)NN * NH * 4);
  bf16_t* tmpA    = (bf16_t*)bump(ws, (size_t)NN * NH * 2);
  bf16_t* hid_bf  = (bf16_t*)bump(ws, (size_t)NN * NH * 2);
  float*  x1      = (float*)bump(ws, (size_t)NN * NH * 4);
  float*  x2      = (float*)bump(ws, (size_t)NN * NH * 4);
  float*  ew      = (float*)bump(ws, (size_t)EE * 4);
  unsigned* key   = (unsigned*)bump(ws, (size_t)EE * 4);
  unsigned* flag  = (unsigned*)bump(ws, (size_t)EE * 4);
  float*  pooled  = (float*)bump(ws, (size_t)NG * NH * 4);
  bf16_t* pool_bf = (bf16_t*)bump(ws, (size_t)NG * NH * 2);
  bf16_t* p_bf    = (bf16_t*)bump(ws, (size_t)NG * NH * 2);
  float*  logits  = (float*)bump(ws, (size_t)NG * 16 * 4);
  bf16_t* Wt_pre  = (bf16_t*)bump(ws, (size_t)NH * NF * 2);
  bf16_t* Wt_c3a  = (bf16_t*)bump(ws, (size_t)NH * NH * 2);
  bf16_t* Wt_c3b  = (bf16_t*)bump(ws, (size_t)NH * NH * 2);
  bf16_t* Wt_c1a  = (bf16_t*)bump(ws, (size_t)NH * NH * 2);
  bf16_t* Wt_c1b  = (bf16_t*)bump(ws, (size_t)NH * NH * 2);
  bf16_t* Wt_c4a  = (bf16_t*)bump(ws, (size_t)NH * NH * 2);
  bf16_t* Wt_c4b  = (bf16_t*)bump(ws, (size_t)NH * NH * 2);
  bf16_t* Wt_post = (bf16_t*)bump(ws, (size_t)NH * NH * 2);
  bf16_t* Wt_ro   = (bf16_t*)bump(ws, (size_t)16 * NH * 2);
  float*  ro_bp   = (float*)bump(ws, 16 * 4);
  unsigned* hist  = (unsigned*)bump(ws, 256 * 4);
  unsigned* state = (unsigned*)bump(ws, 16 * 4);
  unsigned* tcnt  = (unsigned*)bump(ws, TIE_CHUNKS * 4);

  const long nNH   = (long)NN * NH;
  const int  bEW   = (NN * NH + 255) / 256;      // elementwise over N*H
  const int  bEdge = (EE + 7) / 8;               // one wave per edge, 8 per block
  const int  MT    = NN / 16;                    // 3125 row tiles
  const int  gBig  = (MT + 7) / 8;               // 8 row tiles per block (8 waves)
  const int  tchunk = (EE + TIE_CHUNKS - 1) / TIE_CHUNKS;

  // --- weight prep (bf16, transposed [Nout x K]) ---
  k_transpose_bf16<<<(NH * NF + 255) / 256, 256, 0, stream>>>(pre_W, Wt_pre, NF, NH, NH);
  k_transpose_bf16<<<(NH * NH + 255) / 256, 256, 0, stream>>>(c3_W1, Wt_c3a, NH, NH, NH);
  k_transpose_bf16<<<(NH * NH + 255) / 256, 256, 0, stream>>>(c3_W2, Wt_c3b, NH, NH, NH);
  k_transpose_bf16<<<(NH * NH + 255) / 256, 256, 0, stream>>>(c1_W1, Wt_c1a, NH, NH, NH);
  k_transpose_bf16<<<(NH * NH + 255) / 256, 256, 0, stream>>>(c1_W2, Wt_c1b, NH, NH, NH);
  k_transpose_bf16<<<(NH * NH + 255) / 256, 256, 0, stream>>>(c4_W1, Wt_c4a, NH, NH, NH);
  k_transpose_bf16<<<(NH * NH + 255) / 256, 256, 0, stream>>>(c4_W2, Wt_c4b, NH, NH, NH);
  k_transpose_bf16<<<(NH * NH + 255) / 256, 256, 0, stream>>>(post_W, Wt_post, NH, NH, NH);
  k_transpose_bf16<<<(16 * NH + 255) / 256, 256, 0, stream>>>(ro_W, Wt_ro, NH, NCLS, 16);
  k_pad_bias<<<1, 32, 0, stream>>>(ro_b, ro_bp, NCLS, 16);

  // --- h = x @ pre_W + pre_b ---
  k_f32_to_bf16<<<((long)NN * NF + 255) / 256, 256, 0, stream>>>(x, x_bf, (long)NN * NF);
  gemm_wmma_strip<8, NF><<<gBig, 256, 0, stream>>>(x_bf, Wt_pre, pre_b, h, h_bf, MT, 0);

  // --- invariance branch: one GIN conv with unit edge weights (c3, layer 2) ---
  k_zero_f32<<<bEW, 256, 0, stream>>>(agg1, nNH);
  k_scatter_ones<<<bEdge, 256, 0, stream>>>(h, row, col, agg1);
  k_add2h_bf16<<<bEW, 256, 0, stream>>>(agg1, h, tmpA, nNH);
  gemm_wmma_strip<8, NH><<<gBig, 256, 0, stream>>>(tmpA, Wt_c3a, c3_b1, nullptr, hid_bf, MT, 1);
  gemm_wmma_strip<8, NH><<<gBig, 256, 0, stream>>>(hid_bf, Wt_c3b, c3_b2, inv_x, nullptr, MT, 1);

  // --- per-edge dots + sortable keys ---
  k_edge_dot<<<bEdge, 256, 0, stream>>>(inv_x, row, col, ew, key);

  // --- radix select of KTOP-th largest, then exact tie-break flagging ---
  k_sel_init<<<1, 256, 0, stream>>>(state, hist);
  for (int shift = 24; shift >= 0; shift -= 8) {
    k_radix_hist<<<512, 256, 0, stream>>>(key, hist, state, shift);
    k_radix_select<<<1, 1, 0, stream>>>(hist, state);
  }
  k_flag_strict<<<(EE + 255) / 256, 256, 0, stream>>>(key, state, flag);
  k_tie_count<<<(TIE_CHUNKS + 255) / 256, 256, 0, stream>>>(key, state, tcnt, tchunk);
  k_tie_scan<<<1, 1, 0, stream>>>(tcnt);
  k_tie_flag<<<(TIE_CHUNKS + 255) / 256, 256, 0, stream>>>(key, state, tcnt, flag, tchunk);

  // --- split scatter into agg1 (top edges) / agg2 (remaining) ---
  k_zero_f32<<<bEW, 256, 0, stream>>>(agg1, nNH);
  k_zero_f32<<<bEW, 256, 0, stream>>>(agg2, nNH);
  k_scatter_split<<<bEdge, 256, 0, stream>>>(h, row, col, ew, flag, agg1, agg2);

  // --- x1 = relu(GIN(c1, layer 2)) on top edges ---
  k_add2h_bf16<<<bEW, 256, 0, stream>>>(agg1, h, tmpA, nNH);
  gemm_wmma_strip<8, NH><<<gBig, 256, 0, stream>>>(tmpA, Wt_c1a, c1_b1, nullptr, hid_bf, MT, 1);
  gemm_wmma_strip<8, NH><<<gBig, 256, 0, stream>>>(hid_bf, Wt_c1b, c1_b2, x1, nullptr, MT, 1);

  // --- x2 = relu(GIN(c4, layer 2)) on remaining edges ---
  k_add2h_bf16<<<bEW, 256, 0, stream>>>(agg2, h, tmpA, nNH);
  gemm_wmma_strip<8, NH><<<gBig, 256, 0, stream>>>(tmpA, Wt_c4a, c4_b1, nullptr, hid_bf, MT, 1);
  gemm_wmma_strip<8, NH><<<gBig, 256, 0, stream>>>(hid_bf, Wt_c4b, c4_b2, x2, nullptr, MT, 1);

  // --- pooled segment sum, post MLP, readout, log-softmax ---
  k_zero_f32<<<(NG * NH + 255) / 256, 256, 0, stream>>>(pooled, (long)NG * NH);
  k_pool<<<bEW, 256, 0, stream>>>(x1, x2, batch, pooled);
  k_f32_to_bf16<<<(NG * NH + 255) / 256, 256, 0, stream>>>(pooled, pool_bf, (long)NG * NH);
  gemm_wmma_strip<8, NH><<<1, 256, 0, stream>>>(pool_bf, Wt_post, post_b, nullptr, p_bf, NG / 16, 1);
  gemm_wmma_strip<1, NH><<<1, 256, 0, stream>>>(p_bf, Wt_ro, ro_bp, logits, nullptr, NG / 16, 0);
  k_log_softmax<<<1, NG, 0, stream>>>(logits, (float*)d_out);
}
